// Decomposeable_77575699300672
// MI455X (gfx1250) — matrix operand order repaired and checked by reading, not compiled, not run on hardware
//
#include <hip/hip_runtime.h>
#include <hip/hip_bf16.h>
#include <cstdint>

// Problem dims (padded for WMMA tiles)
#define BB   256
#define SS   128
#define VV   32000
#define EE   300
#define EP   320   // E padded to mult of 32
#define E2P  640   // 2E padded
#define PPN  200
#define PPD  224   // P padded to mult of 32
#define P2   448   // 2P padded
#define CC   3

typedef __attribute__((ext_vector_type(16))) _Float16 v16h;
typedef __attribute__((ext_vector_type(8)))  float    v8f;
typedef __attribute__((ext_vector_type(2)))  _Float16 h2;

// ---- WMMA fragment loaders (CDNA5 ISA 7.12.2 layouts) ---------------------
// A 16x32 f16: lane = row (lane&15); K groups 0-7/16-23 (+8 for upper half-wave)
__device__ inline v16h load_fragA(const _Float16* p)
{
    const h2* r = (const h2*)p;                     // 4B-aligned
    h2 q0 = r[0], q1 = r[1], q2 = r[2],  q3 = r[3];
    h2 q4 = r[8], q5 = r[9], q6 = r[10], q7 = r[11];
    v16h f;
    f[0]=q0[0];  f[1]=q0[1];  f[2]=q1[0];  f[3]=q1[1];
    f[4]=q2[0];  f[5]=q2[1];  f[6]=q3[0];  f[7]=q3[1];
    f[8]=q4[0];  f[9]=q4[1];  f[10]=q5[0]; f[11]=q5[1];
    f[12]=q6[0]; f[13]=q6[1]; f[14]=q7[0]; f[15]=q7[1];
    return f;
}
// B 32x16 f16 (stored [N,K]): lane = column (lane&15); K groups 0-7/16-23 (+8 for upper half-wave)
__device__ inline v16h load_fragB(const _Float16* p)
{
    const h2* r = (const h2*)p;
    h2 q0 = r[0], q1 = r[1], q2 = r[2], q3 = r[3];
    h2 q4 = r[8], q5 = r[9], q6 = r[10], q7 = r[11];
    v16h f;
    f[0]=q0[0];  f[1]=q0[1];  f[2]=q1[0];  f[3]=q1[1];
    f[4]=q2[0];  f[5]=q2[1];  f[6]=q3[0];  f[7]=q3[1];
    f[8]=q4[0];  f[9]=q4[1];  f[10]=q5[0]; f[11]=q5[1];
    f[12]=q6[0]; f[13]=q6[1]; f[14]=q7[0]; f[15]=q7[1];
    return f;
}

// ---------------------------------------------------------------------------
// WMMA f16 GEMM:  C[M,N] = (relu?)(A[M,K] @ B^T), B stored [N,K] row-major
// (weights packed transposed; activations transposed by transpose_h), so both
// fragments are contiguous per-lane runs -> global_load_b128 pairs.
// 2x2 register tile: each wave owns a 32x32 C tile -> 4 v_wmma per K-step on
// 8 b128 loads (2x the FLOPs/byte of a 1x1 tile); 4 waves/block along M.
// ---------------------------------------------------------------------------
template<bool RELU, bool OUTF16>
__global__ __launch_bounds__(128)
void gemm_wmma(const _Float16* __restrict__ A, size_t aBatch, int lda,
               const _Float16* __restrict__ Bm, size_t bBatch, int ldb,
               void* __restrict__ Cp, size_t cBatch, int ldc,
               int M, int N, int K)
{
    const int lane = threadIdx.x & 31;
    const int wave = threadIdx.x >> 5;
    const int tN   = blockIdx.x;             // 32-column super-tile
    const int tM   = blockIdx.y * 4 + wave;  // 32-row super-tile (wave-uniform)
    const int bz   = blockIdx.z;
    if (tM * 32 >= M) return;                // uniform: EXEC stays all-ones

    const int halfSel = lane >> 4;           // half-wave K-group select
    const int lan15   = lane & 15;

    const _Float16* A0 = A + bz * aBatch + (size_t)(tM * 32 + lan15) * lda;
    const _Float16* A1 = A0 + (size_t)16 * lda;
    const _Float16* B0 = Bm + bz * bBatch + (size_t)(tN * 32 + lan15) * ldb;
    const _Float16* B1 = B0 + (size_t)16 * ldb;

    v8f acc00 = {}, acc01 = {}, acc10 = {}, acc11 = {};

    for (int kk = 0; kk < K; kk += 32) {
        if (kk + 32 < K) {
            __builtin_prefetch(A0 + kk + 32, 0, 0);
            __builtin_prefetch(A1 + kk + 32, 0, 0);
            __builtin_prefetch(B0 + kk + 32, 0, 0);
            __builtin_prefetch(B1 + kk + 32, 0, 0);
        }
        v16h a0 = load_fragA(A0 + kk + halfSel * 8);
        v16h a1 = load_fragA(A1 + kk + halfSel * 8);
        v16h b0 = load_fragB(B0 + kk + halfSel * 8);
        v16h b1 = load_fragB(B1 + kk + halfSel * 8);
        acc00 = __builtin_amdgcn_wmma_f32_16x16x32_f16(false, a0, false, b0, (short)0, acc00, false, false);
        acc01 = __builtin_amdgcn_wmma_f32_16x16x32_f16(false, a0, false, b1, (short)0, acc01, false, false);
        acc10 = __builtin_amdgcn_wmma_f32_16x16x32_f16(false, a1, false, b0, (short)0, acc10, false, false);
        acc11 = __builtin_amdgcn_wmma_f32_16x16x32_f16(false, a1, false, b1, (short)0, acc11, false, false);
    }

    // ---- epilogue: C VGPR v -> row v (+8 for upper half-wave)
    auto storeAcc = [&](v8f a, int r0, int n0) {
        #pragma unroll
        for (int v = 0; v < 8; ++v) {
            float val = a[v];
            if (RELU) val = fmaxf(val, 0.f);
            size_t idx = bz * cBatch + (size_t)(r0 + halfSel * 8 + v) * ldc + n0 + lan15;
            if (OUTF16) ((_Float16*)Cp)[idx] = (_Float16)val;
            else        ((float*)Cp)[idx]    = val;
        }
    };
    storeAcc(acc00, tM * 32,      tN * 32);
    storeAcc(acc01, tM * 32,      tN * 32 + 16);
    storeAcc(acc10, tM * 32 + 16, tN * 32);
    storeAcc(acc11, tM * 32 + 16, tN * 32 + 16);
}

// ---------------------------------------------------------------------------
// Weight packing, TRANSPOSED: f32 src[nseg*H, C] -> f16 dst[Cp, nseg*Hp]
// (row-major [N,K]); zero-padded. Segments implement the concat-split row
// remap for w_proj / w_cmp.
// ---------------------------------------------------------------------------
__global__ void pack_splitT(const float* __restrict__ src, _Float16* __restrict__ dst,
                            int H, int Hp, int nseg, int C, int Cp)
{
    long t = (long)blockIdx.x * blockDim.x + threadIdx.x;
    long Kp = (long)nseg * Hp;
    long total = Kp * Cp;
    if (t >= total) return;
    int kp = (int)(t % Kp);          // K index (source row, padded)
    int cp = (int)(t / Kp);          // N index (source col, padded)
    int within = kp % Hp;
    int seg    = kp / Hp;
    float val = 0.f;
    if (within < H && cp < C)
        val = src[((size_t)(seg * H + within)) * C + cp];
    dst[t] = (_Float16)val;
}

// ---------------------------------------------------------------------------
// 32x32-tile f16 transpose, src [128, Cn] (lda=ldin) -> dst [Cn, 128].
// Load phase uses gfx1250 async global->LDS copies (ASYNCcnt-tracked),
// drained with s_wait_asynccnt before the barrier; store phase reads LDS
// transposed and emits packed half2 global stores.
// ---------------------------------------------------------------------------
__global__ __launch_bounds__(256)
void transpose_h(const _Float16* __restrict__ src, size_t sBatch, int ldin,
                 _Float16* __restrict__ dst, size_t dBatch, int ldout)
{
    __shared__ _Float16 tile[32][34];            // 34-half row stride: no bank conflicts
    const int b  = blockIdx.z;
    const int c0 = blockIdx.x * 32;              // source-column tile
    const int r0 = blockIdx.y * 32;              // source-row tile
    const _Float16* sb = src + b * sBatch;
    _Float16*       db = dst + b * dBatch;

    const uint32_t ldsBase = (uint32_t)(uintptr_t)(&tile[0][0]);

    // 32 rows x 16 dwords = 512 dwords, 2 per thread, async global->LDS
    #pragma unroll
    for (int p = 0; p < 2; ++p) {
        int idx = threadIdx.x + p * 256;         // 0..511
        int row = idx >> 4;                      // 0..31
        int cw  = idx & 15;                      // dword column (2 halfs)
        uint64_t gaddr = (uint64_t)(uintptr_t)(sb + (size_t)(r0 + row) * ldin + c0 + cw * 2);
        uint32_t loff  = ldsBase + (uint32_t)((row * 34 + cw * 2) * 2);
        asm volatile("global_load_async_to_lds_b32 %0, %1, off"
                     :: "v"(loff), "v"(gaddr) : "memory");
    }
    asm volatile("s_wait_asynccnt 0x0" ::: "memory");
    __syncthreads();

    #pragma unroll
    for (int p = 0; p < 2; ++p) {
        int idx = threadIdx.x + p * 256;
        int c  = idx >> 4;                       // output row (source column) 0..31
        int rw = idx & 15;                       // output dword (2 source rows)
        int r  = rw * 2;
        h2 v;
        v[0] = tile[r][c];
        v[1] = tile[r + 1][c];
        *(h2*)(db + (size_t)(c0 + c) * ldout + r0 + r) = v;
    }
}

// L2-normalize embedding rows, emit f16 rows padded E=300 -> 320.
__global__ void l2norm_emb(const float* __restrict__ emb, _Float16* __restrict__ out)
{
    int row  = blockIdx.x * (blockDim.x >> 5) + (threadIdx.x >> 5);
    int lane = threadIdx.x & 31;
    if (row >= VV) return;
    const float* src = emb + (size_t)row * EE;
    float ss = 0.f;
    for (int i = lane; i < EE; i += 32) { float v = src[i]; ss += v * v; }
    for (int o = 16; o; o >>= 1) ss += __shfl_xor(ss, o, 32);
    float inv = rsqrtf(fmaxf(ss, 1e-12f));
    _Float16* dst = out + (size_t)row * EP;
    for (int i = lane; i < EP; i += 32)
        dst[i] = (i < EE) ? (_Float16)(src[i] * inv) : (_Float16)0.f;
}

// Sequence mask: mask[b,s] = (s < count_nonzero(x[b,:]))
__global__ void seq_mask(const int* __restrict__ x, float* __restrict__ mask)
{
    __shared__ int cnt;
    int b = blockIdx.x, s = threadIdx.x;
    if (s == 0) cnt = 0;
    __syncthreads();
    if (x[b * SS + s] != 0) atomicAdd(&cnt, 1);
    __syncthreads();
    mask[b * SS + s] = (s < cnt) ? 1.f : 0.f;
}

// Gather normalized embedding rows (160 dwords each) into cat[:, :, 0:320].
__global__ void gather_emb(const int* __restrict__ x, const _Float16* __restrict__ embn,
                           _Float16* __restrict__ cat)
{
    long t = (long)blockIdx.x * blockDim.x + threadIdx.x;
    int token = (int)(t / 160);
    int d     = (int)(t % 160);
    if (token >= BB * SS) return;
    int row = x[token];
    ((uint32_t*)(cat + (size_t)token * E2P))[d] =
        ((const uint32_t*)(embn + (size_t)row * EP))[d];
}

// ---- block (128-thread / 4-wave) reductions -------------------------------
__device__ inline float blockMax128(float v, float* red)
{
    for (int o = 16; o; o >>= 1) v = fmaxf(v, __shfl_xor(v, o, 32));
    if ((threadIdx.x & 31) == 0) red[threadIdx.x >> 5] = v;
    __syncthreads();
    float r = fmaxf(fmaxf(red[0], red[1]), fmaxf(red[2], red[3]));
    __syncthreads();
    return r;
}
__device__ inline float blockSum128(float v, float* red)
{
    for (int o = 16; o; o >>= 1) v += __shfl_xor(v, o, 32);
    if ((threadIdx.x & 31) == 0) red[threadIdx.x >> 5] = v;
    __syncthreads();
    float r = red[0] + red[1] + red[2] + red[3];
    __syncthreads();
    return r;
}

// Intra-attention softmax (axis=2) with distance bias (|i-j|>=10).
__global__ void softmax_intra(const float* __restrict__ att, const float* __restrict__ bias,
                              _Float16* __restrict__ sm)
{
    __shared__ float red[4];
    int i = blockIdx.x, b = blockIdx.y, j = threadIdx.x;
    int d = (i > j) ? (i - j) : (j - i);
    float v = att[((size_t)b * SS + i) * SS + j] + ((d >= 10) ? bias[0] : 0.f);
    float m = blockMax128(v, red);
    float e = __expf(v - m);
    float s = blockSum128(e, red);
    sm[((size_t)b * SS + i) * SS + j] = (_Float16)(e / s);
}

// Attend softmax over j (axis=2), multiplicative mask baked in.
__global__ void softmax_att_rows(const float* __restrict__ sim, const float* __restrict__ m1,
                                 const float* __restrict__ m2, _Float16* __restrict__ out)
{
    __shared__ float red[4];
    int i = blockIdx.x, b = blockIdx.y, j = threadIdx.x;
    float v = sim[((size_t)b * SS + i) * SS + j] * m1[b * SS + i] * m2[b * SS + j];
    float m = blockMax128(v, red);
    float e = __expf(v - m);
    float s = blockSum128(e, red);
    out[((size_t)b * SS + i) * SS + j] = (_Float16)(e / s);
}

// Attend softmax over i (axis=1), result written TRANSPOSED: out[b, j, i].
__global__ void softmax_att_colsT(const float* __restrict__ sim, const float* __restrict__ m1,
                                  const float* __restrict__ m2, _Float16* __restrict__ out)
{
    __shared__ float red[4];
    int j = blockIdx.x, b = blockIdx.y, i = threadIdx.x;
    float v = sim[((size_t)b * SS + i) * SS + j] * m1[b * SS + i] * m2[b * SS + j];
    float m = blockMax128(v, red);
    float e = __expf(v - m);
    float s = blockSum128(e, red);
    out[((size_t)b * SS + j) * SS + i] = (_Float16)(e / s);
}

// Masked sum-pool over S: vsum[b, colOff + c] = sum_s v[b,s,c] * mask[b,s]
__global__ void masked_sum(const _Float16* __restrict__ v, const float* __restrict__ mask,
                           float* __restrict__ vsum, int colOff)
{
    int b = blockIdx.x, c = threadIdx.x;       // 448 threads
    const _Float16* vb = v + (size_t)b * SS * P2;
    const float*    mb = mask + b * SS;
    float acc = 0.f;
    for (int s = 0; s < SS; ++s) acc += (float)vb[(size_t)s * P2 + c] * mb[s];
    vsum[(size_t)b * (2 * P2) + colOff + c] = acc;
}

// Final tiny projection: y[b,c] = relu( sum_{k<800} vsum_logical[k] * w_agg[k,c] )
__global__ void final_agg(const float* __restrict__ vsum, const float* __restrict__ w_agg,
                          float* __restrict__ y)
{
    int idx = blockIdx.x * blockDim.x + threadIdx.x;
    if (idx >= BB * CC) return;
    int b = idx / CC, c = idx % CC;
    const float* vs = vsum + (size_t)b * (2 * P2);
    float acc = 0.f;
    for (int k = 0; k < 2 * PPN * 2; ++k) {    // 800
        float vv = (k < 2 * PPN) ? vs[k] : vs[P2 + (k - 2 * PPN)];
        acc += vv * w_agg[(size_t)k * CC + c];
    }
    y[idx] = fmaxf(acc, 0.f);
}

// ---------------------------------------------------------------------------
extern "C" void kernel_launch(void* const* d_in, const int* in_sizes, int n_in,
                              void* d_out, int out_size, void* d_ws, size_t ws_size,
                              hipStream_t stream)
{
    (void)in_sizes; (void)n_in; (void)out_size; (void)ws_size;
    const int*   x1      = (const int*)d_in[0];
    const int*   x2      = (const int*)d_in[1];
    const float* emb     = (const float*)d_in[2];
    const float* w_intra = (const float*)d_in[3];
    const float* b_intra = (const float*)d_in[4];
    const float* w_proj1 = (const float*)d_in[5];
    const float* w_proj2 = (const float*)d_in[6];
    const float* w_att   = (const float*)d_in[7];
    const float* w_cmp1  = (const float*)d_in[8];
    const float* w_cmp2  = (const float*)d_in[9];
    const float* w_agg   = (const float*)d_in[10];
    float*       y       = (float*)d_out;

    // Workspace bump allocator (deterministic layout every call)
    char* base = (char*)d_ws;
    size_t off = 0;
    auto alloc = [&](size_t bytes) -> void* {
        void* p = base + off;
        off = (off + bytes + 255) & ~(size_t)255;
        return p;
    };
    _Float16* wiT   = (_Float16*)alloc((size_t)EP * EP * 2);    // [320,320] (N,K)
    _Float16* wp1T  = (_Float16*)alloc((size_t)PPD * E2P * 2);  // [224,640]
    _Float16* wp2T  = (_Float16*)alloc((size_t)PPD * E2P * 2);
    _Float16* waT   = (_Float16*)alloc((size_t)PPD * PPD * 2);  // [224,224]
    _Float16* wc1T  = (_Float16*)alloc((size_t)P2 * P2 * 2);    // [448,448]
    _Float16* wc2T  = (_Float16*)alloc((size_t)P2 * P2 * 2);
    _Float16* embn  = (_Float16*)alloc((size_t)VV * EP * 2);
    _Float16* cat1  = (_Float16*)alloc((size_t)BB * SS * E2P * 2);
    _Float16* cat2  = (_Float16*)alloc((size_t)BB * SS * E2P * 2);
    _Float16* Fh    = (_Float16*)alloc((size_t)BB * SS * EP * 2);
    _Float16* eT    = (_Float16*)alloc((size_t)BB * EP * SS * 2);   // [B,320,128]
    _Float16* x1pT  = (_Float16*)alloc((size_t)BB * PPD * SS * 2);  // [B,224,128]
    _Float16* x2pT  = (_Float16*)alloc((size_t)BB * PPD * SS * 2);
    float*    attf  = (float*)   alloc((size_t)BB * SS * SS * 4);
    _Float16* smA   = (_Float16*)alloc((size_t)BB * SS * SS * 2);
    _Float16* smB   = (_Float16*)alloc((size_t)BB * SS * SS * 2);
    _Float16* cmp1  = (_Float16*)alloc((size_t)BB * SS * P2 * 2);
    _Float16* cmp2  = (_Float16*)alloc((size_t)BB * SS * P2 * 2);
    _Float16* fb1   = (_Float16*)alloc((size_t)BB * SS * PPD * 2);
    _Float16* fb2   = (_Float16*)alloc((size_t)BB * SS * PPD * 2);
    _Float16* vbuf  = (_Float16*)alloc((size_t)BB * SS * P2 * 2);
    float*    vsum  = (float*)   alloc((size_t)BB * 2 * P2 * 4);
    float*    mask1 = (float*)   alloc((size_t)BB * SS * 4);
    float*    mask2 = (float*)   alloc((size_t)BB * SS * 4);

    dim3 blk(128);
    // 2x2 wave tiles: grid = (N/32, ceil(M/128), batch); 4 waves cover 128 rows
    auto g3 = [](int N_, int M_, int B_) { return dim3(N_ / 32, (M_ + 127) / 128, B_); };
    auto gp = [](long total) { return dim3((unsigned)((total + 255) / 256)); };

    // ---- weight packing (f32 -> transposed, padded f16 [N,K]) ----
    pack_splitT<<<gp((long)EP * EP), 256, 0, stream>>>(w_intra, wiT, EE, EP, 1, EE, EP);
    pack_splitT<<<gp((long)E2P * PPD), 256, 0, stream>>>(w_proj1, wp1T, EE, EP, 2, PPN, PPD);
    pack_splitT<<<gp((long)E2P * PPD), 256, 0, stream>>>(w_proj2, wp2T, EE, EP, 2, PPN, PPD);
    pack_splitT<<<gp((long)PPD * PPD), 256, 0, stream>>>(w_att, waT, PPN, PPD, 1, PPN, PPD);
    pack_splitT<<<gp((long)P2 * P2), 256, 0, stream>>>(w_cmp1, wc1T, PPN, PPD, 2, 2 * PPN, P2);
    pack_splitT<<<gp((long)P2 * P2), 256, 0, stream>>>(w_cmp2, wc2T, PPN, PPD, 2, 2 * PPN, P2);

    // ---- embedding normalize + gather + masks ----
    l2norm_emb<<<VV / 8, 256, 0, stream>>>(emb, embn);
    seq_mask<<<BB, SS, 0, stream>>>(x1, mask1);
    seq_mask<<<BB, SS, 0, stream>>>(x2, mask2);
    gather_emb<<<gp((long)BB * SS * 160), 256, 0, stream>>>(x1, embn, cat1);
    gather_emb<<<gp((long)BB * SS * 160), 256, 0, stream>>>(x2, embn, cat2);

    // ---- intra-attention + project + att-proj, per side ----
    for (int side = 0; side < 2; ++side) {
        _Float16* cat = side ? cat2 : cat1;
        _Float16* wpT = side ? wp2T : wp1T;
        _Float16* cmp = side ? cmp2 : cmp1;
        _Float16* fb  = side ? fb2  : fb1;
        _Float16* xpT = side ? x2pT : x1pT;

        // F = relu(e @ w_intra)            [32768,320]@[320,320]^T-packed
        gemm_wmma<true, true><<<g3(EP, BB * SS, 1), blk, 0, stream>>>(
            cat, 0, E2P, wiT, 0, EP, Fh, 0, EP, BB * SS, EP, EP);
        // att = F F^T (batched)            [B,128,128], K=320
        gemm_wmma<false, false><<<g3(SS, SS, BB), blk, 0, stream>>>(
            Fh, (size_t)SS * EP, EP, Fh, (size_t)SS * EP, EP,
            attf, (size_t)SS * SS, SS, SS, SS, EP);
        // softmax(att + dbias) -> f16
        softmax_intra<<<dim3(SS, BB), 128, 0, stream>>>(attf, b_intra, smA);
        // e^T for xp's B operand (async global->LDS transpose)
        transpose_h<<<dim3(EP / 32, SS / 32, BB), 256, 0, stream>>>(
            cat, (size_t)SS * E2P, E2P, eT, (size_t)EP * SS, SS);
        // xp = sm @ e   -> cat[:, 320:640]
        gemm_wmma<false, true><<<g3(EP, SS, BB), blk, 0, stream>>>(
            smA, (size_t)SS * SS, SS, eT, (size_t)EP * SS, SS,
            (void*)(cat + EP), (size_t)SS * E2P, E2P, SS, EP, SS);
        // xP = relu(cat @ w_proj) -> cmpcat[:, 0:224]
        gemm_wmma<true, true><<<g3(PPD, BB * SS, 1), blk, 0, stream>>>(
            cat, 0, E2P, wpT, 0, E2P, cmp, 0, P2, BB * SS, PPD, E2P);
        // xP^T for alpha/beta B operands
        transpose_h<<<dim3(PPD / 32, SS / 32, BB), 256, 0, stream>>>(
            cmp, (size_t)SS * P2, P2, xpT, (size_t)PPD * SS, SS);
        // f = relu(xP @ w_att)
        gemm_wmma<true, true><<<g3(PPD, BB * SS, 1), blk, 0, stream>>>(
            cmp, 0, P2, waT, 0, PPD, fb, 0, PPD, BB * SS, PPD, PPD);
    }

    // ---- cross attention ----
    // sim = f1 @ f2^T (batched), K=224
    gemm_wmma<false, false><<<g3(SS, SS, BB), blk, 0, stream>>>(
        fb1, (size_t)SS * PPD, PPD, fb2, (size_t)SS * PPD, PPD,
        attf, (size_t)SS * SS, SS, SS, SS, PPD);
    softmax_att_rows <<<dim3(SS, BB), 128, 0, stream>>>(attf, mask1, mask2, smA); // axis=2
    softmax_att_colsT<<<dim3(SS, BB), 128, 0, stream>>>(attf, mask1, mask2, smB); // axis=1, transposed
    // beta  = sm2  @ x2p -> cmp1[:, 224:448]
    gemm_wmma<false, true><<<g3(PPD, SS, BB), blk, 0, stream>>>(
        smA, (size_t)SS * SS, SS, x2pT, (size_t)PPD * SS, SS,
        (void*)(cmp1 + PPD), (size_t)SS * P2, P2, SS, PPD, SS);
    // alpha = sm1T @ x1p -> cmp2[:, 224:448]
    gemm_wmma<false, true><<<g3(PPD, SS, BB), blk, 0, stream>>>(
        smB, (size_t)SS * SS, SS, x1pT, (size_t)PPD * SS, SS,
        (void*)(cmp2 + PPD), (size_t)SS * P2, P2, SS, PPD, SS);

    // ---- compare + masked pooling ----
    gemm_wmma<true, true><<<g3(P2, BB * SS, 1), blk, 0, stream>>>(
        cmp1, 0, P2, wc1T, 0, P2, vbuf, 0, P2, BB * SS, P2, P2);
    masked_sum<<<BB, P2, 0, stream>>>(vbuf, mask1, vsum, 0);
    gemm_wmma<true, true><<<g3(P2, BB * SS, 1), blk, 0, stream>>>(
        cmp2, 0, P2, wc2T, 0, P2, vbuf, 0, P2, BB * SS, P2, P2);
    masked_sum<<<BB, P2, 0, stream>>>(vbuf, mask2, vsum, P2);

    // ---- final projection ----
    final_agg<<<gp(BB * CC), 128, 0, stream>>>(vsum, w_agg, y);
}
